// CausalSelfAttention_9526237462716
// MI455X (gfx1250) — matrix operand order repaired and checked by reading, not compiled, not run on hardware
//
#include <hip/hip_runtime.h>

// ---------------------------------------------------------------------------
// CDNA5 (gfx1250) bf16 WMMA causal self-attention pipeline.
// Shapes: B=2, T=2048, C=2048, H=16, D=128, RD=64.
// ---------------------------------------------------------------------------

typedef __attribute__((ext_vector_type(16))) __bf16 v16bf;
typedef __attribute__((ext_vector_type(8)))  __bf16 v8bf;
typedef __attribute__((ext_vector_type(8)))  float  v8f;
typedef int v4i __attribute__((vector_size(16)));

#define AS1 __attribute__((address_space(1)))
#define AS3 __attribute__((address_space(3)))

typedef const AS1 __bf16* gcbf;   // global const bf16*
typedef AS1 __bf16*       gbf;    // global bf16*
typedef const AS1 v8bf*   gcv8;   // global const v8bf*

// gfx1250 async global->LDS copies (ASYNCcnt-tracked), if this clang has them.
#if defined(__has_builtin)
#if __has_builtin(__builtin_amdgcn_global_load_async_to_lds_b128) && \
    __has_builtin(__builtin_amdgcn_s_wait_asynccnt)
#define USE_ASYNC_LDS 1
#endif
#endif

union Frag { v16bf v; v8bf h[2]; };

__device__ inline v8f vzero() { v8f z; for (int i = 0; i < 8; i++) z[i] = 0.f; return z; }

__device__ inline v8f wmma_bf16(const Frag& a, const Frag& b, v8f c) {
    // D(16x16,f32) = A(16x32,bf16) * B(32x16,bf16) + C
    return __builtin_amdgcn_wmma_f32_16x16x32_bf16(false, a.v, false, b.v,
                                                   (short)0, c, false, false);
}

// ---------------------------------------------------------------------------
// Kernel 0: f32 -> bf16 convert
// ---------------------------------------------------------------------------
__global__ void f32_to_bf16(const float* __restrict__ in, __bf16* __restrict__ out, int n) {
    int i = blockIdx.x * blockDim.x + threadIdx.x;
    if (i < n) out[i] = (__bf16)in[i];
}

// ---------------------------------------------------------------------------
// Kernel 1/4: C(MxN,f32) = A(MxK,bf16 row-major) * B(NxK,bf16 row-major)^T
// Block: 256 threads (8 waves). Block tile 256x128, wave tile 64x64
// (4x4 fragments -> 16 WMMAs per K=32 step against 16x16B loads).
// All global accesses pinned to addrspace(1) so they lower to global_*,
// not flat_* (flat ties up DScnt + the LDS path on CDNA5).
// ---------------------------------------------------------------------------
__global__ __launch_bounds__(256) void gemm_bf16_nt(const __bf16* __restrict__ A,
                                                    const __bf16* __restrict__ B,
                                                    float* __restrict__ C,
                                                    int M, int N, int K) {
    const int lane = threadIdx.x & 31;
    const int w    = threadIdx.x >> 5;
    const int g    = lane >> 4;
    const int nl   = lane & 15;
    const int m0 = blockIdx.x * 256 + (w >> 1) * 64;
    const int n0 = blockIdx.y * 128 + (w & 1) * 64;

    v8f acc[4][4];
    for (int i = 0; i < 4; i++)
        for (int j = 0; j < 4; j++) acc[i][j] = vzero();

    // Steady global pointers; fragment halves sit at +16 (A) / +16 (B) elements.
    gcbf Ag = (gcbf)A;
    gcbf Bg = (gcbf)B;
    gcbf ap[4];
    gcbf bp[4];
    for (int i = 0; i < 4; i++) ap[i] = Ag + (size_t)(m0 + 16 * i + nl) * K + 8 * g;
    for (int j = 0; j < 4; j++) bp[j] = Bg + (size_t)(n0 + 16 * j + nl) * K + 8 * g;

    for (int kb = 0; kb < K; kb += 32) {
        Frag af[4], bfr[4];
        for (int i = 0; i < 4; i++) {
            af[i].h[0] = *(gcv8)(ap[i]);
            af[i].h[1] = *(gcv8)(ap[i] + 16);
            __builtin_prefetch((const __bf16*)(ap[i] + 64), 0, 3);  // 2 iters ahead
            ap[i] += 32;
        }
        for (int j = 0; j < 4; j++) {
            bfr[j].h[0] = *(gcv8)(bp[j]);
            bfr[j].h[1] = *(gcv8)(bp[j] + 16);
            __builtin_prefetch((const __bf16*)(bp[j] + 64), 0, 3);
            bp[j] += 32;
        }
        for (int i = 0; i < 4; i++)
            for (int j = 0; j < 4; j++)
                acc[i][j] = wmma_bf16(af[i], bfr[j], acc[i][j]);
    }
    // C-frag layout: row m = r + 8g, col n = lane&15
    AS1 float* Cg = (AS1 float*)C;
    for (int i = 0; i < 4; i++)
        for (int j = 0; j < 4; j++)
            for (int r = 0; r < 8; r++)
                Cg[(size_t)(m0 + 16 * i + r + 8 * g) * N + (n0 + 16 * j + nl)] = acc[i][j][r];
}

// ---------------------------------------------------------------------------
// Kernel 2: RoPE + head split.
// qkv f32 [B*T, 3C] -> Q,K bf16 [B,H,T,D] (RoPE'd), V^T bf16 [B,H,D,T].
// One thread per (b,t,h,pair). 2^22 threads total.
// ---------------------------------------------------------------------------
__global__ void rope_split(const float* __restrict__ qkv,
                           const float* __restrict__ fcos,
                           const float* __restrict__ fsin,
                           __bf16* __restrict__ Q,
                           __bf16* __restrict__ K,
                           __bf16* __restrict__ VT) {
    int tid = blockIdx.x * blockDim.x + threadIdx.x;  // 22 bits
    int p  = tid & 63;           // pair index 0..63
    int hh = (tid >> 6) & 15;    // head
    int t  = (tid >> 10) & 2047; // time
    int b  = tid >> 21;          // batch

    size_t row = (size_t)(b * 2048 + t) * 6144;
    int col = hh * 128 + 2 * p;

    float qr = qkv[row + col],        qi = qkv[row + col + 1];
    float kr = qkv[row + 2048 + col], ki = qkv[row + 2048 + col + 1];
    float vr = qkv[row + 4096 + col], vi = qkv[row + 4096 + col + 1];
    float c = fcos[t * 64 + p], s = fsin[t * 64 + p];

    size_t ho = ((size_t)(b * 16 + hh) * 2048 + t) * 128 + 2 * p;
    Q[ho]     = (__bf16)(qr * c - qi * s);
    Q[ho + 1] = (__bf16)(qr * s + qi * c);
    K[ho]     = (__bf16)(kr * c - ki * s);
    K[ho + 1] = (__bf16)(kr * s + ki * c);

    size_t vo = ((size_t)(b * 16 + hh) * 128 + 2 * p) * 2048 + t;
    VT[vo]        = (__bf16)vr;
    VT[vo + 2048] = (__bf16)vi;
}

// ---------------------------------------------------------------------------
// Kernel 3: causal flash attention, one (b,h,64-q-row tile) per block.
// 128 threads = 4 waves; wave w owns q rows q0+16w .. q0+16w+15.
// Keys processed 32 at a time, staged in LDS (async copy when available);
// online softmax; y -> bf16 [B,T,C].
// ---------------------------------------------------------------------------
__global__ __launch_bounds__(128) void flash_attn(const __bf16* __restrict__ Q,
                                                  const __bf16* __restrict__ Kh,
                                                  const __bf16* __restrict__ VT,
                                                  __bf16* __restrict__ Y) {
    const int T = 2048, D = 128;
    const int tid  = threadIdx.x;
    const int lane = tid & 31;
    const int w    = tid >> 5;
    const int g    = lane >> 4;
    const int nl   = lane & 15;
    const int bh = blockIdx.x >> 5;       // b*16 + h
    const int q0 = (blockIdx.x & 31) * 64;
    const int b  = bh >> 4;
    const int hh = bh & 15;

    __shared__ __align__(16) __bf16 Kt[32 * 128];     // [key][d]   8 KB
    __shared__ __align__(16) __bf16 Vt[128 * 32];     // [d][key]   8 KB
    __shared__ __align__(16) __bf16 Pt[4][16 * 32];   // per-wave P 4 KB

    gcbf Qg  = (gcbf)Q;
    gcbf Kg  = (gcbf)Kh;
    gcbf VTg = (gcbf)VT;

    // Preload Q A-fragments (16 rows x 128 d -> 4 K=32 fragments)
    Frag qf[4];
    {
        gcbf qrow = Qg + ((size_t)bh * T + (q0 + 16 * w + nl)) * D;
        for (int dk = 0; dk < 4; dk++) {
            qf[dk].h[0] = *(gcv8)(qrow + 32 * dk + 8 * g);
            qf[dk].h[1] = *(gcv8)(qrow + 32 * dk + 16 + 8 * g);
        }
    }

    v8f O[8];
    for (int i = 0; i < 8; i++) O[i] = vzero();
    float mrun[8], lrun[8];
    for (int r = 0; r < 8; r++) { mrun[r] = -1e30f; lrun[r] = 0.f; }
    const float sc = 0.08838834764831845f;  // 1/sqrt(128)

    for (int kt0 = 0; kt0 < q0 + 64; kt0 += 32) {
        __syncthreads();
        // Stage K tile (32x128, contiguous in global since t-stride == D)
        // and V^T tile Vt[d][0..31] from VT[bh][d][kt0..kt0+31].
        {
            gcbf ksrc = Kg + ((size_t)bh * T + kt0) * D;
#ifdef USE_ASYNC_LDS
            for (int j = 0; j < 4; j++) {
                int c = tid + j * 128;
                __builtin_amdgcn_global_load_async_to_lds_b128(
                    (AS1 v4i*)(ksrc + c * 8), (AS3 v4i*)(Kt + c * 8), 0, 0);
            }
            for (int j = 0; j < 4; j++) {
                int c = tid + j * 128;
                int d = c >> 2, part = c & 3;
                __builtin_amdgcn_global_load_async_to_lds_b128(
                    (AS1 v4i*)(VTg + ((size_t)bh * D + d) * T + kt0 + part * 8),
                    (AS3 v4i*)(Vt + d * 32 + part * 8), 0, 0);
            }
            __builtin_amdgcn_s_wait_asynccnt(0);
#else
            for (int j = 0; j < 4; j++) {
                int c = tid + j * 128;
                *(v8bf*)(Kt + c * 8) = *(gcv8)(ksrc + c * 8);
            }
            for (int j = 0; j < 4; j++) {
                int c = tid + j * 128;
                int d = c >> 2, part = c & 3;
                *(v8bf*)(Vt + d * 32 + part * 8) =
                    *(gcv8)(VTg + ((size_t)bh * D + d) * T + kt0 + part * 8);
            }
#endif
        }
        __syncthreads();

        // S = Q * K^T for two 16-key halves (K-dim = d, 4 steps of 32)
        v8f s0 = vzero(), s1 = vzero();
        for (int nh = 0; nh < 2; nh++) {
            v8f s = vzero();
            int key = 16 * nh + nl;  // B-frag: n = key, table-order k = d
            for (int dk = 0; dk < 4; dk++) {
                Frag kf;
                kf.h[0] = *(const v8bf*)(Kt + key * 128 + 32 * dk + 8 * g);
                kf.h[1] = *(const v8bf*)(Kt + key * 128 + 32 * dk + 8 * g + 16);
                s = wmma_bf16(qf[dk], kf, s);
            }
            if (nh == 0) s0 = s; else s1 = s;
        }

        // Online softmax: each lane holds rows m = r+8g at col n = lane&15;
        // row stats reduced across the 16-lane halves with shfl_xor.
        for (int r = 0; r < 8; r++) {
            int mg = q0 + 16 * w + r + 8 * g;  // global q row
            float a0 = (kt0 + nl      <= mg) ? s0[r] * sc : -1e30f;
            float a1 = (kt0 + 16 + nl <= mg) ? s1[r] * sc : -1e30f;
            float mx = fmaxf(a0, a1);
            for (int msk = 1; msk < 16; msk <<= 1) mx = fmaxf(mx, __shfl_xor(mx, msk, 32));
            float mn = fmaxf(mrun[r], mx);
            float rs = __expf(mrun[r] - mn);
            float p0 = __expf(a0 - mn);
            float p1 = __expf(a1 - mn);
            float ps = p0 + p1;
            for (int msk = 1; msk < 16; msk <<= 1) ps += __shfl_xor(ps, msk, 32);
            lrun[r] = lrun[r] * rs + ps;
            mrun[r] = mn;
            for (int dt = 0; dt < 8; dt++) O[dt][r] *= rs;
            // Transpose P through LDS (C-layout -> A-layout)
            Pt[w][(r + 8 * g) * 32 + nl]      = (__bf16)p0;
            Pt[w][(r + 8 * g) * 32 + 16 + nl] = (__bf16)p1;
        }

        // O += P(16x32) * V(32x128): reload P as A-fragment, Vt as B-fragments
        Frag pf;
        pf.h[0] = *(const v8bf*)(&Pt[w][nl * 32 + 8 * g]);
        pf.h[1] = *(const v8bf*)(&Pt[w][nl * 32 + 16 + 8 * g]);
        for (int dt = 0; dt < 8; dt++) {
            Frag vf;
            vf.h[0] = *(const v8bf*)(Vt + (16 * dt + nl) * 32 + 8 * g);
            vf.h[1] = *(const v8bf*)(Vt + (16 * dt + nl) * 32 + 8 * g + 16);
            O[dt] = wmma_bf16(pf, vf, O[dt]);
        }
    }

    // Normalize and emit y as bf16 in [B,T,C] (feeds out-proj GEMM A operand)
    for (int r = 0; r < 8; r++) {
        float inv = 1.0f / lrun[r];
        int row = q0 + 16 * w + r + 8 * g;
        gbf yrow = (gbf)Y + ((size_t)(b * T + row)) * 2048 + hh * D;
        for (int dt = 0; dt < 8; dt++)
            yrow[16 * dt + nl] = (__bf16)(O[dt][r] * inv);
    }
}

// ---------------------------------------------------------------------------
// Launch
// ---------------------------------------------------------------------------
extern "C" void kernel_launch(void* const* d_in, const int* in_sizes, int n_in,
                              void* d_out, int out_size, void* d_ws, size_t ws_size,
                              hipStream_t stream) {
    const float* x  = (const float*)d_in[0];
    // d_in[1]: mask (bool) — causal structure applied analytically
    const float* fc = (const float*)d_in[2];
    const float* fs = (const float*)d_in[3];
    const float* wa = (const float*)d_in[4];
    const float* wp = (const float*)d_in[5];
    float* out = (float*)d_out;

    char* ws = (char*)d_ws;
    __bf16* xb  = (__bf16*)(ws);                 // 16.8 MB  x  bf16 [4096,2048]
    __bf16* wab = (__bf16*)(ws + 16777216);      // 25.2 MB  w_attn bf16 [6144,2048]
    __bf16* wpb = (__bf16*)(ws + 41943040);      //  8.4 MB  w_proj bf16 [2048,2048]
    float*  qkv = (float*)(ws + 50331648);       // 100.7 MB qkv f32 [4096,6144]
    __bf16* Qb  = (__bf16*)(ws + 150994944);     // 16.8 MB  Q bf16 [B,H,T,D]
    __bf16* Kb  = (__bf16*)(ws + 167772160);     // 16.8 MB  K bf16 [B,H,T,D]
    __bf16* VTb = (__bf16*)(ws + 184549376);     // 16.8 MB  V^T bf16 [B,H,D,T]
    __bf16* yb  = (__bf16*)(ws + 201326592);     // 16.8 MB  y bf16 [4096,2048]

    f32_to_bf16<<<8388608 / 256, 256, 0, stream>>>(x, xb, 8388608);
    f32_to_bf16<<<12582912 / 256, 256, 0, stream>>>(wa, wab, 12582912);
    f32_to_bf16<<<4194304 / 256, 256, 0, stream>>>(wp, wpb, 4194304);

    // qkv = x @ w_attn^T   (M=4096, N=6144, K=2048), block tile 256x128
    gemm_bf16_nt<<<dim3(16, 48), 256, 0, stream>>>(xb, wab, qkv, 4096, 6144, 2048);

    // RoPE + split into head-major Q/K and transposed V
    rope_split<<<16384, 256, 0, stream>>>(qkv, fc, fs, Qb, Kb, VTb);

    // Causal flash attention: B*H*(T/64) = 1024 blocks, 4 waves each
    flash_attn<<<1024, 128, 0, stream>>>(Qb, Kb, VTb, yb);

    // out = y @ w_proj^T   (M=4096, N=2048, K=2048)
    gemm_bf16_nt<<<dim3(16, 16), 256, 0, stream>>>(yb, wpb, out, 4096, 2048, 2048);

    (void)in_sizes; (void)n_in; (void)out_size; (void)ws_size;
}
